// ViewAndScenePoint2Global_33088428049086
// MI455X (gfx1250) — matrix operand order repaired and checked by reading, not compiled, not run on hardware
//
#include <hip/hip_runtime.h>
#include <hip/hip_bf16.h>

typedef __attribute__((ext_vector_type(16))) __bf16   v16bf;
typedef __attribute__((ext_vector_type(8)))  float    v8f;
typedef __attribute__((ext_vector_type(4)))  float    v4f;
typedef __attribute__((ext_vector_type(4)))  unsigned v4u;

#define NHEAD 8

// round-to-nearest-even float -> bf16 bits
static __device__ __forceinline__ unsigned short f2bf(float f) {
  union { float f; unsigned u; } v; v.f = f;
  unsigned r = v.u + 0x7FFFu + ((v.u >> 16) & 1u);
  return (unsigned short)(r >> 16);
}

static __device__ __forceinline__ v16bf combine(v4u lo, v4u hi) {
  union { v4u q[2]; v16bf b; } u; u.q[0] = lo; u.q[1] = hi; return u.b;
}

// Issue the two ds_load_b128 for one 32-byte B fragment at LDS byte offset
// `addr` (wave-relative). asm volatile => never hoisted/CSE'd, always DS path.
static __device__ __forceinline__ void ldB_issue(unsigned addr, v4u& lo, v4u& hi) {
  asm volatile("ds_load_b128 %0, %2\n\t"
               "ds_load_b128 %1, %2 offset:16"
               : "=v"(lo), "=v"(hi)
               : "v"(addr));
}
// Wait until at most the 2 most-recently-issued DS ops are outstanding
// (i.e. the tied fragment's loads are done). Tied "+v" operands force the
// consuming WMMA to stay below the wait.
static __device__ __forceinline__ void waitB2(v4u& lo, v4u& hi) {
  asm volatile("s_wait_dscnt 0x2" : "+v"(lo), "+v"(hi));
}
static __device__ __forceinline__ void waitB0(v4u& lo, v4u& hi) {
  asm volatile("s_wait_dscnt 0x0" : "+v"(lo), "+v"(hi));
}

// ---------------------------------------------------------------------------
// Convert 256x256 fp32 weights (row-major [K][N]) into bf16 pre-swizzled into
// this kernel's contiguous-k B-operand (32x16 KxN, 16-bit) per-lane layout:
//   fragment (n, kk): lane = (N%16) + 16*((K%32)/16), element e = K%16
//   dst offset (bf16 units) = ((n*8 + kk)*32 + lane)*16 + e
// so each lane's 16 bf16 elements are 32 contiguous bytes.
// ---------------------------------------------------------------------------
__global__ void swizzle_w_bf16(const float* __restrict__ W,
                               unsigned short* __restrict__ D) {
  int i = blockIdx.x * blockDim.x + threadIdx.x;
  for (; i < 256 * 256; i += gridDim.x * blockDim.x) {
    const int K = i >> 8, N = i & 255;
    const int n = N >> 4, ln = N & 15;
    const int kk = K >> 5, kr = K & 31;
    const int lane = ln + ((kr >> 4) << 4);
    const int e = kr & 15;
    const int off = (((n * 8 + kk) * 32 + lane) << 4) + e;
    D[off] = f2bf(W[(K << 8) + N]);
  }
}

// ---------------------------------------------------------------------------
// xr_v = (relu(LN(prev)) @ g2v_W + g2v_b) @ v_Wr + v_br   [256]; xr_s likewise.
// One block, 512 threads.
// ---------------------------------------------------------------------------
__global__ void prep_xr(const float* __restrict__ prev,
                        const float* __restrict__ g2v_s, const float* __restrict__ g2v_b,
                        const float* __restrict__ g2vW,  const float* __restrict__ g2vbb,
                        const float* __restrict__ g2s_s, const float* __restrict__ g2s_b,
                        const float* __restrict__ g2sW,  const float* __restrict__ g2sbb,
                        const float* __restrict__ vWr,   const float* __restrict__ vbr,
                        const float* __restrict__ sWr,   const float* __restrict__ sbr,
                        float* __restrict__ xrv, float* __restrict__ xrs) {
  __shared__ float tv[512], ts[512], xtv[256], xts[256];
  __shared__ float r1[512], r2[512];
  const int tid = threadIdx.x;  // 512 threads
  const float x = prev[tid];
  r1[tid] = x; r2[tid] = x * x;
  __syncthreads();
  for (int s = 256; s > 0; s >>= 1) {
    if (tid < s) { r1[tid] += r1[tid + s]; r2[tid] += r2[tid + s]; }
    __syncthreads();
  }
  const float mean = r1[0] * (1.f / 512.f);
  const float var  = r2[0] * (1.f / 512.f) - mean * mean;
  const float inv  = rsqrtf(var + 1e-5f);
  const float nx   = (x - mean) * inv;
  float a = nx * g2v_s[tid] + g2v_b[tid]; tv[tid] = a > 0.f ? a : 0.f;
  float b = nx * g2s_s[tid] + g2s_b[tid]; ts[tid] = b > 0.f ? b : 0.f;
  __syncthreads();
  if (tid < 256) {
    float acc = g2vbb[tid];
    for (int k = 0; k < 512; ++k) acc += tv[k] * g2vW[k * 256 + tid];
    xtv[tid] = acc;
  } else {
    const int c = tid - 256;
    float acc = g2sbb[c];
    for (int k = 0; k < 512; ++k) acc += ts[k] * g2sW[k * 256 + c];
    xts[c] = acc;
  }
  __syncthreads();
  if (tid < 256) {
    float acc = vbr[tid];
    for (int k = 0; k < 256; ++k) acc += xtv[k] * vWr[k * 256 + tid];
    xrv[tid] = acc;
  } else {
    const int c = tid - 256;
    float acc = sbr[c];
    for (int k = 0; k < 256; ++k) acc += xts[k] * sWr[k * 256 + c];
    xrs[c] = acc;
  }
}

// One pipelined fragment step: issue next fragment's loads, wait on current,
// WMMA-accumulate, then a scheduling fence so the scheduler cannot sink the
// WMMA (extending fragment live ranges into spills).
#define GAT_STEP(J, CL, CH, NL, NH_, CACC, AI)                                 \
  {                                                                            \
    if ((J) < 15) ldB_issue(hbase + ((J) + 1) * 1024u, NL, NH_);               \
    if ((J) < 15) waitB2(CL, CH); else waitB0(CL, CH);                         \
    CACC = __builtin_amdgcn_wmma_f32_16x16x32_bf16(                            \
        false, Afr[AI], false, combine(CL, CH), (short)0, CACC, false, false); \
    __builtin_amdgcn_sched_barrier(0);                                         \
  }

// ---------------------------------------------------------------------------
// Main streaming kernel: per wave, process 16-row tiles of X.
//   xl_tile[16x256] = X_tile @ Wl + bl  via v_wmma_f32_16x16x32_bf16
// then per head (two 16-col fragments): e = leaky_relu(xl + xr),
// logits = <e, att>, online softmax update of (m, s, acc[256]).
// Weights staged once per block into 128KB LDS; B fragments streamed from LDS
// via hand-pipelined inline-asm ds_load_b128 pairs (double buffered).
// Partial per wave: [0:8]=m, [8:16]=s, [16:272]=acc.
// ---------------------------------------------------------------------------
__global__ void __launch_bounds__(256) gat_stream(
    const float* __restrict__ X, int numTiles,
    const unsigned short* __restrict__ Wswz,
    const float* __restrict__ bl,
    const float* __restrict__ xr,
    const float* __restrict__ att,
    float* __restrict__ partials, int totalWaves) {
  __shared__ unsigned short ldsW[256 * 256];  // 128 KB swizzled bf16 weights
  __shared__ float ldsBl[256], ldsXr[256], ldsAtt[256];
  const int tid  = threadIdx.x;
  const int lane = tid & 31;
  const int wave = tid >> 5;
  {
    const uint4* src = (const uint4*)Wswz;
    uint4* dst = (uint4*)ldsW;
    for (int i = tid; i < 8192; i += 256) dst[i] = src[i];
    ldsBl[tid] = bl[tid]; ldsXr[tid] = xr[tid]; ldsAtt[tid] = att[tid];
  }
  __syncthreads();

  const int lr = lane & 15;   // column within fragment / row within A half
  const int hi = lane >> 4;   // lane half selector
  const int gw = blockIdx.x * 8 + wave;
  // 32-bit wave-relative LDS byte offset of this lane's B-fragment slot
  const unsigned wb = ((unsigned)(uintptr_t)(&ldsW[0])) + (unsigned)(lane * 32);

  // Hoist per-lane bias / xr / att scalars into registers (fragment-indexed):
  // fragment f covers columns [16f, 16f+16), this lane's column is 16f+lr.
  float blr[16], xrr[16], atr[16];
#pragma unroll
  for (int f = 0; f < 16; ++f) {
    blr[f] = ldsBl[16 * f + lr];
    xrr[f] = ldsXr[16 * f + lr];
    atr[f] = ldsAtt[16 * f + lr];
  }

  float mh[NHEAD], sh[NHEAD], accf[16];
#pragma unroll
  for (int h = 0; h < NHEAD; ++h) { mh[h] = -__builtin_inff(); sh[h] = 0.f; }
#pragma unroll
  for (int f = 0; f < 16; ++f) accf[f] = 0.f;

  for (int t = gw; t < numTiles; t += totalWaves) {
    // ---- load + convert A fragments (16x32 bf16 each), contiguous-k A layout (same k order as B):
    //      K = 16*(lane>=16) + e, row M = lane%16
    const float* rp = X + (size_t)(t * 16 + lr) * 256;
    v16bf Afr[8];
#pragma unroll
    for (int kk = 0; kk < 8; ++kk) {
      const int k0 = kk * 32 + hi * 16;
      const v4f f0 = *(const v4f*)(rp + k0);
      const v4f f1 = *(const v4f*)(rp + k0 + 4);
      const v4f f2 = *(const v4f*)(rp + k0 + 8);
      const v4f f3 = *(const v4f*)(rp + k0 + 12);
      union { unsigned short u[16]; v16bf b; } cv;
      cv.u[0]  = f2bf(f0.x); cv.u[1]  = f2bf(f0.y); cv.u[2]  = f2bf(f0.z); cv.u[3]  = f2bf(f0.w);
      cv.u[4]  = f2bf(f1.x); cv.u[5]  = f2bf(f1.y); cv.u[6]  = f2bf(f1.z); cv.u[7]  = f2bf(f1.w);
      cv.u[8]  = f2bf(f2.x); cv.u[9]  = f2bf(f2.y); cv.u[10] = f2bf(f2.z); cv.u[11] = f2bf(f2.w);
      cv.u[12] = f2bf(f3.x); cv.u[13] = f2bf(f3.y); cv.u[14] = f2bf(f3.z); cv.u[15] = f2bf(f3.w);
      Afr[kk] = cv.b;
    }

#pragma unroll
    for (int h = 0; h < NHEAD; ++h) {
      // fragment j=0..15 of this head: n = 2h + (j>=8), kk = j%8
      // LDS byte offset = wb + h*16384 + j*1024
      const unsigned hbase = wb + (unsigned)(h * 16384);
      v8f c0 = {}; v8f c1 = {};
      v4u b0l, b0h, b1l, b1h;
      ldB_issue(hbase, b0l, b0h);
      GAT_STEP(0,  b0l, b0h, b1l, b1h, c0, 0)
      GAT_STEP(1,  b1l, b1h, b0l, b0h, c0, 1)
      GAT_STEP(2,  b0l, b0h, b1l, b1h, c0, 2)
      GAT_STEP(3,  b1l, b1h, b0l, b0h, c0, 3)
      GAT_STEP(4,  b0l, b0h, b1l, b1h, c0, 4)
      GAT_STEP(5,  b1l, b1h, b0l, b0h, c0, 5)
      GAT_STEP(6,  b0l, b0h, b1l, b1h, c0, 6)
      GAT_STEP(7,  b1l, b1h, b0l, b0h, c0, 7)
      GAT_STEP(8,  b0l, b0h, b1l, b1h, c1, 0)
      GAT_STEP(9,  b1l, b1h, b0l, b0h, c1, 1)
      GAT_STEP(10, b0l, b0h, b1l, b1h, c1, 2)
      GAT_STEP(11, b1l, b1h, b0l, b0h, c1, 3)
      GAT_STEP(12, b0l, b0h, b1l, b1h, c1, 4)
      GAT_STEP(13, b1l, b1h, b0l, b0h, c1, 5)
      GAT_STEP(14, b0l, b0h, b1l, b1h, c1, 6)
      GAT_STEP(15, b1l, b1h, b0l, b0h, c1, 7)

      // C fragment layout: lane l -> col = l%16, VGPR v -> row v + 8*(l>=16)
      const float bl0 = blr[2 * h],  bl1 = blr[2 * h + 1];
      const float xr0 = xrr[2 * h],  xr1 = xrr[2 * h + 1];
      const float at0 = atr[2 * h],  at1 = atr[2 * h + 1];
      float q[8];
#pragma unroll
      for (int v = 0; v < 8; ++v) {
        float x0 = c0[v] + bl0; c0[v] = x0;  // xl (with bias) for output einsum
        float x1 = c1[v] + bl1; c1[v] = x1;
        float a0 = x0 + xr0; a0 = a0 > 0.f ? a0 : 0.2f * a0;  // leaky_relu 0.2
        float a1 = x1 + xr1; a1 = a1 > 0.f ? a1 : 0.2f * a1;
        q[v] = a0 * at0 + a1 * at1;          // per-column logit contribution
      }
      // sum over 16 columns (lanes within each half-wave)
#pragma unroll
      for (int v = 0; v < 8; ++v) {
        q[v] += __shfl_xor(q[v], 1);
        q[v] += __shfl_xor(q[v], 2);
        q[v] += __shfl_xor(q[v], 4);
        q[v] += __shfl_xor(q[v], 8);
      }
      float mt = q[0];
#pragma unroll
      for (int v = 1; v < 8; ++v) mt = fmaxf(mt, q[v]);
      mt = fmaxf(mt, __shfl_xor(mt, 16));            // max over all 16 rows
      const float mnew = fmaxf(mh[h], mt);
      const float corr = __expf(mh[h] - mnew);
      float swl = 0.f, A0 = 0.f, A1 = 0.f;
#pragma unroll
      for (int v = 0; v < 8; ++v) {
        const float w = __expf(q[v] - mnew);
        swl += w; A0 += w * c0[v]; A1 += w * c1[v];
      }
      swl += __shfl_xor(swl, 16);
      A0  += __shfl_xor(A0, 16);
      A1  += __shfl_xor(A1, 16);
      sh[h] = sh[h] * corr + swl;
      mh[h] = mnew;
      accf[2 * h]     = accf[2 * h]     * corr + A0;
      accf[2 * h + 1] = accf[2 * h + 1] * corr + A1;
    }
  }

  float* P = partials + (size_t)gw * 272;
  if (lane == 0) {
#pragma unroll
    for (int h = 0; h < NHEAD; ++h) { P[h] = mh[h]; P[8 + h] = sh[h]; }
  }
  if (lane < 16) {
#pragma unroll
    for (int f = 0; f < 16; ++f) P[16 + f * 16 + lane] = accf[f];
  }
}

// ---------------------------------------------------------------------------
// Final: merge partials (LSE combine), residual, LN, relu, 512x512 MLP, skip.
// One block, 256 threads.
// ---------------------------------------------------------------------------
__global__ void reduce_final(
    const float* __restrict__ PV, int nV,
    const float* __restrict__ PS, int nS,
    const float* __restrict__ vbias, const float* __restrict__ sbias,
    const float* __restrict__ prev,
    const float* __restrict__ plns, const float* __restrict__ plnb,
    const float* __restrict__ mlpW, const float* __restrict__ mlpb,
    float* __restrict__ out) {
  __shared__ float xs[512], tn[512], r1[256], r2[256];
  const int tid = threadIdx.x;  // 256
  {
    const int c = tid, h = c >> 5;
    float M = -__builtin_inff();
    for (int p = 0; p < nV; ++p) M = fmaxf(M, PV[p * 272 + h]);
    float S = 0.f, Acc = 0.f;
    for (int p = 0; p < nV; ++p) {
      const float w = __expf(PV[p * 272 + h] - M);
      S   += w * PV[p * 272 + 8 + h];
      Acc += w * PV[p * 272 + 16 + c];
    }
    xs[c] = prev[c] + Acc / S + vbias[c];
  }
  {
    const int c = tid, h = c >> 5;
    float M = -__builtin_inff();
    for (int p = 0; p < nS; ++p) M = fmaxf(M, PS[p * 272 + h]);
    float S = 0.f, Acc = 0.f;
    for (int p = 0; p < nS; ++p) {
      const float w = __expf(PS[p * 272 + h] - M);
      S   += w * PS[p * 272 + 8 + h];
      Acc += w * PS[p * 272 + 16 + c];
    }
    xs[256 + c] = prev[256 + c] + Acc / S + sbias[c];
  }
  __syncthreads();
  const float a = xs[tid], b = xs[tid + 256];
  r1[tid] = a + b; r2[tid] = a * a + b * b;
  __syncthreads();
  for (int s = 128; s > 0; s >>= 1) {
    if (tid < s) { r1[tid] += r1[tid + s]; r2[tid] += r2[tid + s]; }
    __syncthreads();
  }
  const float mean = r1[0] * (1.f / 512.f);
  const float var  = r2[0] * (1.f / 512.f) - mean * mean;
  const float inv  = rsqrtf(var + 1e-5f);
  {
    float u = (a - mean) * inv * plns[tid] + plnb[tid];
    tn[tid] = u > 0.f ? u : 0.f;
    float w2 = (b - mean) * inv * plns[tid + 256] + plnb[tid + 256];
    tn[tid + 256] = w2 > 0.f ? w2 : 0.f;
  }
  __syncthreads();
  for (int jj = 0; jj < 2; ++jj) {
    const int j = tid + jj * 256;
    float r = mlpb[j];
    for (int k = 0; k < 512; ++k) r += tn[k] * mlpW[k * 512 + j];
    out[j] = xs[j] + r;
  }
}

extern "C" void kernel_launch(void* const* d_in, const int* in_sizes, int n_in,
                              void* d_out, int out_size, void* d_ws, size_t ws_size,
                              hipStream_t stream) {
  (void)in_sizes; (void)n_in; (void)out_size; (void)ws_size;
  const float* view  = (const float*)d_in[0];
  const float* scene = (const float*)d_in[1];
  const float* prev  = (const float*)d_in[2];
  const float* g2v_s = (const float*)d_in[3];
  const float* g2v_b = (const float*)d_in[4];
  const float* g2v_W = (const float*)d_in[5];
  const float* g2v_bb= (const float*)d_in[6];
  const float* g2s_s = (const float*)d_in[7];
  const float* g2s_b = (const float*)d_in[8];
  const float* g2s_W = (const float*)d_in[9];
  const float* g2s_bb= (const float*)d_in[10];
  const float* vWl  = (const float*)d_in[11];
  const float* vbl  = (const float*)d_in[12];
  const float* vWr  = (const float*)d_in[13];
  const float* vbr  = (const float*)d_in[14];
  const float* vatt = (const float*)d_in[15];
  const float* vbias= (const float*)d_in[16];
  const float* sWl  = (const float*)d_in[17];
  const float* sbl  = (const float*)d_in[18];
  const float* sWr  = (const float*)d_in[19];
  const float* sbr  = (const float*)d_in[20];
  const float* satt = (const float*)d_in[21];
  const float* sbias= (const float*)d_in[22];
  const float* plns = (const float*)d_in[23];
  const float* plnb = (const float*)d_in[24];
  const float* mlpW = (const float*)d_in[25];
  const float* mlpb = (const float*)d_in[26];
  float* out = (float*)d_out;

  char* ws = (char*)d_ws;
  float* xrv          = (float*)(ws + 0);
  float* xrs          = (float*)(ws + 1024);
  unsigned short* WV  = (unsigned short*)(ws + 2048);
  unsigned short* WS  = (unsigned short*)(ws + 2048 + 131072);
  float* PV           = (float*)(ws + 264192);            // 256 partials
  float* PS           = (float*)(ws + 264192 + 278528);   // 2048 partials

  swizzle_w_bf16<<<64, 256, 0, stream>>>(vWl, WV);
  swizzle_w_bf16<<<64, 256, 0, stream>>>(sWl, WS);
  prep_xr<<<1, 512, 0, stream>>>(prev, g2v_s, g2v_b, g2v_W, g2v_bb,
                                 g2s_s, g2s_b, g2s_W, g2s_bb,
                                 vWr, vbr, sWr, sbr, xrv, xrs);
  const int nTilesV = 4096 / 16;   const int wavesV = 32 * 8;    // 1 tile/wave
  const int nTilesS = 262144 / 16; const int wavesS = 256 * 8;   // 8 tiles/wave
  gat_stream<<<32, 256, 0, stream>>>(view, nTilesV, WV, vbl, xrv, vatt, PV, wavesV);
  gat_stream<<<256, 256, 0, stream>>>(scene, nTilesS, WS, sbl, xrs, satt, PS, wavesS);
  reduce_final<<<1, 256, 0, stream>>>(PV, wavesV, PS, wavesS, vbias, sbias, prev,
                                      plns, plnb, mlpW, mlpb, out);
}
